// MyGCN_38397007626981
// MI455X (gfx1250) — compile-verified, matrix operand order checked by reading, not run on hardware
//
#include <hip/hip_runtime.h>
#include <hip/hip_bf16.h>

// ---------------------------------------------------------------------------
// 3-layer GCN + global attention pooling for MI455X (gfx1250, wave32).
// Dense transforms use v_wmma_f32_16x16x32_f16 (fp16 in, fp32 accumulate).
// Graph gather/scatter is L2-resident (feature matrix = 25.6 MB << 192 MB L2)
// and uses hardware fp32 global atomics.
// N = 100000 = 6250 * 16  -> every WMMA tile is full, no guards needed.
// ---------------------------------------------------------------------------

#define N_NODES   100000
#define N_EDGES   1250000
#define EDGE_FEAT 13
#define HID       64

typedef __attribute__((ext_vector_type(16))) _Float16 v16h;
typedef __attribute__((ext_vector_type(8)))  _Float16 v8h;
typedef __attribute__((ext_vector_type(8)))  float    v8f;

__device__ __forceinline__ float warpSum32(float v) {
#pragma unroll
    for (int m = 16; m >= 1; m >>= 1) v += __shfl_xor(v, m, 32);
    return v;
}

// ---------------------------------------------------------------------------
// coef = softmax(aaaaa) over 13 entries (trivial, single lane)
// ---------------------------------------------------------------------------
__global__ void coef_kernel(const float* __restrict__ a, float* __restrict__ coef) {
    if (threadIdx.x == 0) {
        float mx = -1e30f;
        for (int j = 0; j < EDGE_FEAT; ++j) mx = fmaxf(mx, a[j]);
        float e[EDGE_FEAT], s = 0.f;
        for (int j = 0; j < EDGE_FEAT; ++j) { e[j] = __expf(a[j] - mx); s += e[j]; }
        float inv = 1.f / s;
        for (int j = 0; j < EDGE_FEAT; ++j) coef[j] = e[j] * inv;
    }
}

// zero deg[N] and the 3 pooling accumulators (3*65 floats)
__global__ void zero_kernel(float* __restrict__ deg, float* __restrict__ pools) {
    int i = blockIdx.x * blockDim.x + threadIdx.x;
    if (i < N_NODES) deg[i] = 0.f;
    if (i < 3 * 65) pools[i] = 0.f;
}

// ew[e] = edge_attr[e,:] . coef ; deg[col[e]] += ew[e]
__global__ void edge_weight_deg(const float* __restrict__ attr,
                                const int*   __restrict__ ei,
                                const float* __restrict__ coef,
                                float* __restrict__ ew,
                                float* __restrict__ deg) {
    int e = blockIdx.x * blockDim.x + threadIdx.x;
    if (e >= N_EDGES) return;
    const float* ar = attr + (long)e * EDGE_FEAT;
    float w = 0.f;
#pragma unroll
    for (int j = 0; j < EDGE_FEAT; ++j) w += ar[j] * coef[j];
    ew[e] = w;
    int col = ei[(long)N_EDGES + e];
    unsafeAtomicAdd(&deg[col], w);
}

// dinv[i] = rsqrt(deg[i]+1)  (self loop weight 1 -> deg always >= 1)
// selfnorm[i] = dinv[i]^2    (the self-loop edge's norm)
__global__ void dinv_kernel(const float* __restrict__ deg,
                            float* __restrict__ dinv,
                            float* __restrict__ selfnorm) {
    int i = blockIdx.x * blockDim.x + threadIdx.x;
    if (i >= N_NODES) return;
    float dv = rsqrtf(deg[i] + 1.0f);
    dinv[i] = dv;
    selfnorm[i] = dv * dv;
}

// norm[e] = dinv[row]*ew[e]*dinv[col]
__global__ void norm_kernel(const float* __restrict__ ew,
                            const float* __restrict__ dinv,
                            const int*   __restrict__ ei,
                            float* __restrict__ norm) {
    int e = blockIdx.x * blockDim.x + threadIdx.x;
    if (e >= N_EDGES) return;
    norm[e] = dinv[ei[e]] * ew[e] * dinv[ei[(long)N_EDGES + e]];
}

// ---------------------------------------------------------------------------
// H = X @ W  via WMMA f32_16x16x32_f16.
// Grid: 625 blocks x 320 threads (10 waves); wave w of block b owns rows
// [ (b*10+w)*16, +16 )  -> 625*10*16 == 100000 exactly, no partial tiles.
// W (64x64) staged in LDS as f16, transposed to wlds[n*64+k] so each lane's
// B fragment is two contiguous 8-half runs (ds_load_b128).
// 16-bit A layout (ISA 7.12.2): lane m (+16*half) holds row m; slots 0..7 are
// K = 8*half + 0..7, slots 8..15 are K = 16 + 8*half + 0..7 (per 32-K chunk),
// i.e. two contiguous float8 runs of the source row -> float4 loads.
// All 32 output stores share one base address + immediate offsets.
// ---------------------------------------------------------------------------
__global__ void __launch_bounds__(320) gemm64_wmma(const float* __restrict__ X,
                                                   const float* __restrict__ W,
                                                   float* __restrict__ H) {
    __shared__ _Float16 wlds[64 * 64]; // transposed: wlds[n*64 + k]
    const int tid = threadIdx.x;
    for (int idx = tid; idx < 64 * 64; idx += 320) {
        int k = idx >> 6, n = idx & 63;
        wlds[n * 64 + k] = (_Float16)W[idx];
    }
    __syncthreads();

    const int wave = tid >> 5;
    const int lane = tid & 31;
    const int half = lane >> 4;
    const int m    = lane & 15;
    const int tileRow = (blockIdx.x * 10 + wave) * 16; // always full tile

    // A fragments for the two K-chunks
    v16h afrag[2];
    {
        const float* xr = X + (long)(tileRow + m) * HID;
#pragma unroll
        for (int kc = 0; kc < 2; ++kc) {
            const int base = kc * 32 + half * 8;
            const float4 lo0 = *(const float4*)(xr + base + 0);
            const float4 lo1 = *(const float4*)(xr + base + 4);
            const float4 hi0 = *(const float4*)(xr + base + 16);
            const float4 hi1 = *(const float4*)(xr + base + 20);
            afrag[kc][0]  = (_Float16)lo0.x; afrag[kc][1]  = (_Float16)lo0.y;
            afrag[kc][2]  = (_Float16)lo0.z; afrag[kc][3]  = (_Float16)lo0.w;
            afrag[kc][4]  = (_Float16)lo1.x; afrag[kc][5]  = (_Float16)lo1.y;
            afrag[kc][6]  = (_Float16)lo1.z; afrag[kc][7]  = (_Float16)lo1.w;
            afrag[kc][8]  = (_Float16)hi0.x; afrag[kc][9]  = (_Float16)hi0.y;
            afrag[kc][10] = (_Float16)hi0.z; afrag[kc][11] = (_Float16)hi0.w;
            afrag[kc][12] = (_Float16)hi1.x; afrag[kc][13] = (_Float16)hi1.y;
            afrag[kc][14] = (_Float16)hi1.z; afrag[kc][15] = (_Float16)hi1.w;
        }
    }

    // Single base pointer; all stores become immediate-offset global_store_b32.
    float* hp = H + (long)(tileRow + half * 8) * HID + m;

#pragma unroll
    for (int nt = 0; nt < 4; ++nt) {
        v8f acc = {};
#pragma unroll
        for (int kc = 0; kc < 2; ++kc) {
            const int base = kc * 32 + half * 8;
            const _Float16* wp = &wlds[(nt * 16 + m) * 64 + base];
            v8h blo = *(const v8h*)(wp);       // K = base..base+7
            v8h bhi = *(const v8h*)(wp + 16);  // K = base+16..base+23
            v16h bfrag;
#pragma unroll
            for (int i = 0; i < 8; ++i) { bfrag[i] = blo[i]; bfrag[8 + i] = bhi[i]; }
            acc = __builtin_amdgcn_wmma_f32_16x16x32_f16(
                false, afrag[kc], false, bfrag, (short)0, acc, false, false);
        }
        // C/D layout: VGPR rr -> row (rr + 8*half), col = nt*16 + m
#pragma unroll
        for (int rr = 0; rr < 8; ++rr) {
            hp[rr * HID + nt * 16] = acc[rr];  // immediate offsets, unguarded
        }
    }
}

// out[i][f] = selfnorm[i] * h[i][f]   (self-loop contribution seeds the accumulator)
__global__ void init_out(const float* __restrict__ h,
                         const float* __restrict__ selfnorm,
                         float* __restrict__ out) {
    long i = (long)blockIdx.x * blockDim.x + threadIdx.x;
    if (i < (long)N_NODES * HID) out[i] = selfnorm[i >> 6] * h[i];
}

// One wave per edge: out[col] += norm[e] * h[row]  (2 x 128B coalesced, L2 atomics)
__global__ void __launch_bounds__(256) scatter_edges(const float* __restrict__ h,
                                                     const float* __restrict__ norm,
                                                     const int*   __restrict__ ei,
                                                     float* __restrict__ out) {
    int e    = blockIdx.x * (blockDim.x >> 5) + (threadIdx.x >> 5);
    int lane = threadIdx.x & 31;
    if (e >= N_EDGES) return;
    int row = ei[e];
    int col = ei[(long)N_EDGES + e];
    float nv = norm[e];
    const float* hr = h + (long)row * HID;
    float v0 = nv * hr[lane];
    float v1 = nv * hr[32 + lane];
    float* orow = out + (long)col * HID;
    unsafeAtomicAdd(&orow[lane], v0);
    unsafeAtomicAdd(&orow[32 + lane], v1);
}

// x = relu(x + b) in place; accumulate attention-pool numerator/denominator.
// gate = sigmoid(x.wg + bg) in (0,1) -> exp(gate) bounded, single-pass softmax.
// Wave handles 4 nodes; block reduces into LDS then 65 global atomics/block.
__global__ void __launch_bounds__(256) bias_relu_pool(float* __restrict__ xio,
                                                      const float* __restrict__ b,
                                                      const float* __restrict__ wg,
                                                      const float* __restrict__ bg,
                                                      float* __restrict__ pool) {
    __shared__ float lacc[65];
    const int tid  = threadIdx.x;
    const int wave = tid >> 5;
    const int lane = tid & 31;
    if (tid < 65) lacc[tid] = 0.f;
    __syncthreads();

    const float wg0 = wg[lane],      wg1 = wg[32 + lane];
    const float b0  = b[lane],       b1  = b[32 + lane];
    const float bgv = bg[0];
    float pn0 = 0.f, pn1 = 0.f, pd = 0.f;

#pragma unroll
    for (int k = 0; k < 4; ++k) {
        int node = blockIdx.x * 32 + wave * 4 + k; // 3125*32 == 100000 exactly
        float* xr = xio + (long)node * HID;
        float v0 = fmaxf(xr[lane] + b0, 0.f);
        float v1 = fmaxf(xr[32 + lane] + b1, 0.f);
        xr[lane] = v0;
        xr[32 + lane] = v1;
        float z = warpSum32(v0 * wg0 + v1 * wg1) + bgv;
        float gate = 1.f / (1.f + __expf(-z));
        float eg = __expf(gate);
        pn0 += eg * v0;
        pn1 += eg * v1;
        pd  += eg;
    }
    atomicAdd(&lacc[lane], pn0);
    atomicAdd(&lacc[32 + lane], pn1);
    if (lane == 0) atomicAdd(&lacc[64], pd);
    __syncthreads();
    if (tid < 65) unsafeAtomicAdd(&pool[tid], lacc[tid]);
}

// out[p*64+f] = pools[p][f] / pools[p][64]
__global__ void finalize(const float* __restrict__ pools, float* __restrict__ out) {
    int t = threadIdx.x;
    if (t < 3 * HID) {
        int p = t / HID, f = t % HID;
        const float* pp = pools + p * 65;
        out[t] = pp[f] / pp[64];
    }
}

// ---------------------------------------------------------------------------
extern "C" void kernel_launch(void* const* d_in, const int* in_sizes, int n_in,
                              void* d_out, int out_size, void* d_ws, size_t ws_size,
                              hipStream_t stream) {
    const float* x      = (const float*)d_in[0];
    const int*   ei     = (const int*)d_in[1];   // edge_index [2, E] row-major
    const float* attr   = (const float*)d_in[2];
    const float* aaaaa  = (const float*)d_in[3];
    const float* Wl[3]  = { (const float*)d_in[4], (const float*)d_in[6], (const float*)d_in[8] };
    const float* bl[3]  = { (const float*)d_in[5], (const float*)d_in[7], (const float*)d_in[9] };
    const float* wgl[3] = { (const float*)d_in[10], (const float*)d_in[12], (const float*)d_in[14] };
    const float* bgl[3] = { (const float*)d_in[11], (const float*)d_in[13], (const float*)d_in[15] };
    float* out = (float*)d_out;

    // Workspace layout (floats, 64B-aligned slices)
    float* ws = (float*)d_ws;
    const long E = N_EDGES, N = N_NODES;
    float* coef     = ws;                       // 16
    float* ew       = coef + 16;                // E
    float* deg      = ew + E;                   // N
    float* dinv     = deg + N;                  // N
    float* selfnorm = dinv + N;                 // N
    float* nrm      = selfnorm + N;             // E
    float* h        = nrm + E;                  // N*64
    float* xa       = h + N * HID;              // N*64
    float* xb       = xa + N * HID;             // N*64
    float* pools    = xb + N * HID;             // 3*65
    (void)ws_size; (void)n_in; (void)in_sizes; (void)out_size;

    const int T = 256;
    zero_kernel<<<(N_NODES + T - 1) / T, T, 0, stream>>>(deg, pools);
    coef_kernel<<<1, 32, 0, stream>>>(aaaaa, coef);
    edge_weight_deg<<<(N_EDGES + T - 1) / T, T, 0, stream>>>(attr, ei, coef, ew, deg);
    dinv_kernel<<<(N_NODES + T - 1) / T, T, 0, stream>>>(deg, dinv, selfnorm);
    norm_kernel<<<(N_EDGES + T - 1) / T, T, 0, stream>>>(ew, dinv, ei, nrm);

    const float* xin[3]  = { x, xa, xb };
    float*       xout[3] = { xa, xb, xa };
    const int gemmBlocks     = 625;                 // 625*10 waves * 16 rows = 100000
    const int elemBlocks     = (N_NODES * HID) / T; // 25000 exact
    const int edgeWaveBlocks = N_EDGES / 8;         // 156250 exact, 1 edge/wave
    const int poolBlocks     = N_NODES / 32;        // 3125 exact

    for (int L = 0; L < 3; ++L) {
        gemm64_wmma<<<gemmBlocks, 320, 0, stream>>>(xin[L], Wl[L], h);
        init_out<<<elemBlocks, T, 0, stream>>>(h, selfnorm, xout[L]);
        scatter_edges<<<edgeWaveBlocks, T, 0, stream>>>(h, nrm, ei, xout[L]);
        bias_relu_pool<<<poolBlocks, T, 0, stream>>>(xout[L], bl[L], wgl[L], bgl[L],
                                                     pools + L * 65);
    }
    finalize<<<1, 192, 0, stream>>>(pools, out);
}
